// EdgeHeterophilyScorer_11123965296914
// MI455X (gfx1250) — compile-verified, matrix operand order checked by reading, not run on hardware
//
#include <hip/hip_runtime.h>
#include <hip/hip_bf16.h>

#define B_N 1024
#define FEAT 512
#define HDIM 256
#define TAU 5.0f
#define EPS_V 1e-8f

typedef __attribute__((ext_vector_type(16))) __bf16 v16bf;
typedef __attribute__((ext_vector_type(8)))  float  v8f;

union Frag {
    v16bf v;
    unsigned int u[8];
    uint4 u4[2];
};

__device__ __forceinline__ unsigned int f2bf(float f) {
    unsigned int u = __float_as_uint(f);
    unsigned int r = u + 0x7FFFu + ((u >> 16) & 1u);   // RNE to bf16
    return r >> 16;
}

// |a-b| on packed bf16 pairs: native packed subtract (exact + RNE) then abs mask.
__device__ __forceinline__ unsigned int absdiff_pair(unsigned int a, unsigned int b) {
    unsigned int d;
    asm("v_pk_add_bf16 %0, %1, %2 neg_lo:[0,1] neg_hi:[0,1]"
        : "=v"(d) : "v"(a), "v"(b));
    return d & 0x7FFF7FFFu;
}

// Branchless erf (Abramowitz-Stegun 7.1.26, |err| < 1.5e-7): v_rcp + v_exp, no exec masking
__device__ __forceinline__ float erf_fast(float x) {
    float ax = fabsf(x);
    float t  = __builtin_amdgcn_rcpf(fmaf(0.3275911f, ax, 1.0f));
    float p  = fmaf(t, 1.061405429f, -1.453152027f);
    p = fmaf(t, p, 1.421413741f);
    p = fmaf(t, p, -0.284496736f);
    p = fmaf(t, p, 0.254829592f);
    float r = 1.0f - p * t * __expf(-ax * ax);
    return copysignf(r, x);
}

__device__ __forceinline__ float gelu_exact(float v) {
    return 0.5f * v * (1.0f + erf_fast(v * 0.70710678f));
}

// ---------------- kernel 1a: class softmax (C=2) ----------------
__global__ void prep_p_kernel(const float* __restrict__ logits, float* __restrict__ p) {
    int i = blockIdx.x * blockDim.x + threadIdx.x;
    if (i < B_N) {
        float a = logits[2 * i], b = logits[2 * i + 1];
        float m = fmaxf(a, b);
        float ea = __expf(a - m), eb = __expf(b - m);
        float inv = __builtin_amdgcn_rcpf(ea + eb);
        p[2 * i] = ea * inv;
        p[2 * i + 1] = eb * inv;
    }
}

// ---------------- kernel 1b: pre-swizzle W1[:512] into WMMA B-fragment layout ----
// out dword index = (((nt*16)+kt)*32 + lane)*8 + s   (16 n-tiles, 16 k-tiles)
__global__ void pack_w1_kernel(const float* __restrict__ W1, unsigned int* __restrict__ out) {
    int gid = blockIdx.x * blockDim.x + threadIdx.x;   // 65536 dwords
    int s    = gid & 7;
    int lane = (gid >> 3) & 31;
    int kt   = (gid >> 8) & 15;
    int nt   = (gid >> 12) & 15;
    int half = lane >> 4;
    int N    = nt * 16 + (lane & 15);
    int k0   = kt * 32 + ((s < 4) ? (half * 8 + 2 * s) : (16 + half * 8 + 2 * (s - 4)));
    unsigned int lo = f2bf(W1[k0 * HDIM + N]);
    unsigned int hi = f2bf(W1[(k0 + 1) * HDIM + N]);
    out[gid] = lo | (hi << 16);
}

// ---------------- kernel 2: fused implicit GEMM + epilogue ----------------
__global__ __launch_bounds__(256) void edge_scorer_kernel(
    const float* __restrict__ Xc,          // [1024,512]
    const float* __restrict__ adj,         // [1024,1024]
    const float* __restrict__ W1,          // [514,256] (rows 512/513 used here)
    const float* __restrict__ b1,          // [256]
    const float* __restrict__ W2,          // [256]
    const float* __restrict__ b2,          // [1]
    const float* __restrict__ p,           // [1024,2] softmaxed labels
    const unsigned int* __restrict__ pW,   // packed W1 fragments
    float* __restrict__ out)               // [0..B*B)=logits(->adj_refined), [B*B..2B*B)=h_scores
{
    __shared__ unsigned int Xi[16][256];   // 16 i-rows, 512 bf16 packed (16 KB)
    __shared__ unsigned int Xj[16][256];   // 16 j-rows (16 KB)
    __shared__ float cB1[256], cW5a[256], cW5b[256], cW2[256];
    __shared__ float pjL[16][2];
    __shared__ float scoreAcc[256];

    const int tid  = threadIdx.x;
    const int wave = tid >> 5;
    const int lane = tid & 31;
    const int half = lane >> 4;
    const int mcol = lane & 15;
    const int i0 = blockIdx.y * 16;
    const int j0 = blockIdx.x * 16;

    // ---- stage X rows as packed bf16, consts, zero score accumulator ----
    for (int idx = tid; idx < 4096; idx += 256) {
        int t = idx >> 8, d = idx & 255;
        float2 gi = *(const float2*)&Xc[(i0 + t) * FEAT + 2 * d];
        float2 gj = *(const float2*)&Xc[(j0 + t) * FEAT + 2 * d];
        Xi[t][d] = f2bf(gi.x) | (f2bf(gi.y) << 16);
        Xj[t][d] = f2bf(gj.x) | (f2bf(gj.y) << 16);
    }
    cB1[tid]  = b1[tid];
    cW5a[tid] = W1[512 * HDIM + tid];
    cW5b[tid] = W1[513 * HDIM + tid];
    cW2[tid]  = W2[tid];
    if (tid < 32) pjL[tid >> 1][tid & 1] = p[(j0 + (tid >> 1)) * 2 + (tid & 1)];
    scoreAcc[tid] = 0.0f;
    __syncthreads();

    // issue 4 B-fragments (one n-group) for tile (pass, kt)
    auto loadB4 = [&](Frag* Bf, int pass, int kt) {
#pragma unroll
        for (int n = 0; n < 4; ++n) {
            int nt = pass * 4 + n;
            const uint4* q = (const uint4*)(pW + (((nt * 16) + kt) * 32 + lane) * 8);
            Bf[n].u4[0] = q[0];
            Bf[n].u4[1] = q[1];
        }
    };
    // build bf16 A fragments for both M-tiles (ISA 16-bit A 16x32 layout)
    auto buildA2 = [&](Frag* A, int kt) {
        int d0 = kt * 16 + half * 4;        // K run {0..7}/{8..15}
        int d1 = kt * 16 + 8 + half * 4;    // K run {16..23}/{24..31}
        uint4 xj0 = *(const uint4*)&Xj[mcol][d0];
        uint4 xj1 = *(const uint4*)&Xj[mcol][d1];
#pragma unroll
        for (int a = 0; a < 2; ++a) {
            int t = 2 * wave + a;
            uint4 xi0 = *(const uint4*)&Xi[t][d0];
            uint4 xi1 = *(const uint4*)&Xi[t][d1];
            A[a].u[0] = absdiff_pair(xi0.x, xj0.x);
            A[a].u[1] = absdiff_pair(xi0.y, xj0.y);
            A[a].u[2] = absdiff_pair(xi0.z, xj0.z);
            A[a].u[3] = absdiff_pair(xi0.w, xj0.w);
            A[a].u[4] = absdiff_pair(xi1.x, xj1.x);
            A[a].u[5] = absdiff_pair(xi1.y, xj1.y);
            A[a].u[6] = absdiff_pair(xi1.z, xj1.z);
            A[a].u[7] = absdiff_pair(xi1.w, xj1.w);
        }
    };

    // per-lane epilogue constants
    float pj0r[8], pj1r[8];
#pragma unroll
    for (int r = 0; r < 8; ++r) {
        pj0r[r] = pjL[r + 8 * half][0];
        pj1r[r] = pjL[r + 8 * half][1];
    }
    float pi0[2], pi1[2];
#pragma unroll
    for (int a = 0; a < 2; ++a) {
        int i = i0 + 2 * wave + a;
        pi0[a] = p[2 * i];
        pi1[a] = p[2 * i + 1];
    }

    float sPart[2][8];
#pragma unroll
    for (int a = 0; a < 2; ++a)
#pragma unroll
        for (int r = 0; r < 8; ++r) sPart[a][r] = 0.0f;

    // ---- GEMM: wave owns M-tiles {2w,2w+1}; 4 passes of 4 N-tiles; K = 512.
    // B-fragments are ping-pong double-buffered across kt so WMMAs overlap loads.
    Frag B0[4], B1[4], A[2];
    loadB4(B0, 0, 0);
    for (int pass = 0; pass < 4; ++pass) {
        v8f acc[2][4] = {};
        for (int kt = 0; kt < 16; kt += 2) {
            // prefetch kt+1 while computing kt
            loadB4(B1, pass, kt + 1);
            buildA2(A, kt);
#pragma unroll
            for (int n = 0; n < 4; ++n) {
                acc[0][n] = __builtin_amdgcn_wmma_f32_16x16x32_bf16(
                    false, A[0].v, false, B0[n].v, (short)0, acc[0][n], false, false);
                acc[1][n] = __builtin_amdgcn_wmma_f32_16x16x32_bf16(
                    false, A[1].v, false, B0[n].v, (short)0, acc[1][n], false, false);
            }
            // prefetch kt+2 (or next pass's kt=0 at the boundary; clamped in-bounds)
            int np = (kt + 2 == 16) ? ((pass + 1) & 3) : pass;
            loadB4(B0, np, (kt + 2) & 15);
            buildA2(A, kt + 1);
#pragma unroll
            for (int n = 0; n < 4; ++n) {
                acc[0][n] = __builtin_amdgcn_wmma_f32_16x16x32_bf16(
                    false, A[0].v, false, B1[n].v, (short)0, acc[0][n], false, false);
                acc[1][n] = __builtin_amdgcn_wmma_f32_16x16x32_bf16(
                    false, A[1].v, false, B1[n].v, (short)0, acc[1][n], false, false);
            }
        }
        // ---- fused epilogue: bias + rank-2 label term + exact GELU + dot(W2) ----
#pragma unroll
        for (int n = 0; n < 4; ++n) {
            int nG = (pass * 4 + n) * 16 + mcol;   // C/D layout: N = lane&15
            float bb = cB1[nG], w5a = cW5a[nG], w5b = cW5b[nG], w2n = cW2[nG];
#pragma unroll
            for (int a = 0; a < 2; ++a) {
#pragma unroll
                for (int r = 0; r < 8; ++r) {      // C/D layout: M = r + 8*half
                    float v = acc[a][n][r] + bb
                            + pi0[a] * pj0r[r] * w5a
                            + pi1[a] * pj1r[r] * w5b;
                    sPart[a][r] += gelu_exact(v) * w2n;
                }
            }
        }
    }

    // cross-lane N reduction into per-pair score (ds_add_f32)
#pragma unroll
    for (int a = 0; a < 2; ++a)
#pragma unroll
        for (int r = 0; r < 8; ++r)
            atomicAdd(&scoreAcc[(2 * wave + a) * 16 + (r + 8 * half)], sPart[a][r]);
    __syncthreads();

    // ---- per-pair finish: sigmoid, h_scores out, softmax logits in-place ----
    {
        int t = tid >> 4, m = tid & 15;
        int i = i0 + t, j = j0 + m;
        float sc = scoreAcc[tid] + b2[0];
        float s = __builtin_amdgcn_rcpf(1.0f + __expf(-sc));
        out[B_N * B_N + i * B_N + j] = s;                              // h_scores
        out[i * B_N + j] = adj[i * B_N + j] + TAU * __logf(s + EPS_V); // softmax logits
    }
}

// ---------------- kernel 3: in-place row softmax of adj_refined ----------------
__global__ void row_softmax_kernel(float* __restrict__ x) {
    __shared__ float red[256];
    const int tid = threadIdx.x;
    float* row = x + (size_t)blockIdx.x * B_N;

    float m = -3.4e38f;
    for (int c = tid; c < B_N; c += 256) m = fmaxf(m, row[c]);
    red[tid] = m; __syncthreads();
    for (int off = 128; off > 0; off >>= 1) {
        if (tid < off) red[tid] = fmaxf(red[tid], red[tid + off]);
        __syncthreads();
    }
    m = red[0]; __syncthreads();

    float s = 0.0f;
    for (int c = tid; c < B_N; c += 256) s += __expf(row[c] - m);
    red[tid] = s; __syncthreads();
    for (int off = 128; off > 0; off >>= 1) {
        if (tid < off) red[tid] += red[tid + off];
        __syncthreads();
    }
    float inv = 1.0f / red[0]; __syncthreads();

    for (int c = tid; c < B_N; c += 256) row[c] = __expf(row[c] - m) * inv;
}

extern "C" void kernel_launch(void* const* d_in, const int* in_sizes, int n_in,
                              void* d_out, int out_size, void* d_ws, size_t ws_size,
                              hipStream_t stream) {
    (void)in_sizes; (void)n_in; (void)out_size; (void)ws_size;
    const float* Xc   = (const float*)d_in[0];
    const float* lgt  = (const float*)d_in[1];
    const float* adj  = (const float*)d_in[2];
    const float* W1   = (const float*)d_in[3];
    const float* b1   = (const float*)d_in[4];
    const float* W2   = (const float*)d_in[5];
    const float* b2   = (const float*)d_in[6];
    float* out = (float*)d_out;

    float* ws_p = (float*)d_ws;                           // 2048 floats
    unsigned int* ws_pw = (unsigned int*)(ws_p + 2048);   // 65536 dwords (256 KB)

    prep_p_kernel<<<dim3(4), dim3(256), 0, stream>>>(lgt, ws_p);
    pack_w1_kernel<<<dim3(256), dim3(256), 0, stream>>>(W1, ws_pw);
    edge_scorer_kernel<<<dim3(64, 64), dim3(256), 0, stream>>>(
        Xc, adj, W1, b1, W2, b2, ws_p, ws_pw, out);
    row_softmax_kernel<<<dim3(1024), dim3(256), 0, stream>>>(out);
}
